// MixHopNetwork_59090160058610
// MI455X (gfx1250) — matrix operand order repaired and checked by reading, not compiled, not run on hardware
//
#include <hip/hip_runtime.h>
#include <stdint.h>

// MixHop forward for MI455X (gfx1250, wave32, WMMA).
// Heavy GEMMs: bf16 V_WMMA_F32_16X16X32_BF16 with f32 accumulation.
// adj (400MB f32) converted once to bf16 (~200MB, ~fits 192MB L2), swept 4x
// with batched column blocks (800/400/800/400): ~0.8GB adj traffic for
// 480 GFLOP -> near the bf16-WMMA / 23.3TB/s balance point.
// Data movement: global_load_async_to_lds_b128 (ASYNCcnt) -> LDS double
// buffer -> ds_load_b128 fragments -> WMMA. All tiles are in-bounds by
// construction (padded buffers), so the loader has zero branches/guards.

#define N_NODES 10000
#define F_IN    500
#define F_PAD   512
#define H_DIM   400
#define C_DIM   1200
#define C_PAD   1216
#define M_PAD   10112   // 79 * 128
#define K_ADJ   10016   // 10000 rounded up to 32 (zero-padded reduction cols)
#define NW_PAD  1280    // weight-T row allocation (10 N-tiles)
#define UV_ROWS 1296    // U^T/V^T rows: slice +400 read through tile 896
#define TS_ROWS 896     // T1^T/S1^T rows (7 N-tiles)

typedef __attribute__((ext_vector_type(16))) __bf16 v16bf;
typedef __attribute__((ext_vector_type(8)))  __bf16 v8bf;
typedef __attribute__((ext_vector_type(8)))  float  v8f;
typedef __attribute__((ext_vector_type(8)))  unsigned short u16x8;

__device__ __forceinline__ unsigned short bf16_rne(float f) {
  union { float f; uint32_t u; } cv; cv.f = f;
  uint32_t u = cv.u;
  if ((u & 0x7fffffffu) > 0x7f800000u) return (unsigned short)((u >> 16) | 0x0040u); // NaN
  return (unsigned short)((u + 0x7fffu + ((u >> 16) & 1u)) >> 16);
}

// ---- packing kernels -------------------------------------------------------

// dst[row][0:dstCols] = bf16(src[row][0:srcCols]) zero-padded. grid.y = row.
__global__ void k_cvt_pad(const float* __restrict__ src,
                          unsigned short* __restrict__ dst,
                          int srcCols, int dstCols) {
  const long row = blockIdx.y;
  const int  c   = blockIdx.x * 256 + threadIdx.x;
  if (c >= dstCols) return;
  float v = (c < srcCols) ? src[row * (long)srcCols + c] : 0.f;
  dst[row * (long)dstCols + c] = bf16_rne(v);
}

// dst[n][k] (n<3*Hdim, k<ldk) = W_{n/Hdim}[k][n%Hdim], zero-padded in k.
__global__ void k_pack_w3T(const float* __restrict__ W0, const float* __restrict__ W1,
                           const float* __restrict__ W2, int Kdim, int Hdim,
                           unsigned short* __restrict__ dst, int ldk, long total) {
  long i = (long)blockIdx.x * 256 + threadIdx.x;
  if (i >= total) return;
  int n = (int)(i / ldk), k = (int)(i % ldk);
  const float* W = (n < Hdim) ? W0 : (n < 2 * Hdim) ? W1 : W2;
  int nn = n % Hdim;
  float v = (k < Kdim) ? W[(long)k * Hdim + nn] : 0.f;
  dst[i] = bf16_rne(v);
}

// dst[n][k] = W_fc[k][n] (1200x1200 -> [1200][1216], zero-padded in k).
__global__ void k_pack_wfcT(const float* __restrict__ W,
                            unsigned short* __restrict__ dst, long total) {
  long i = (long)blockIdx.x * 256 + threadIdx.x;
  if (i >= total) return;
  int n = (int)(i / C_PAD), k = (int)(i % C_PAD);
  float v = (k < C_DIM) ? W[(long)k * C_DIM + n] : 0.f;
  dst[i] = bf16_rne(v);
}

// zero columns [c0, ld) of a [rows][ld] bf16 buffer (K-padding of reductions)
__global__ void k_zero_pad(unsigned short* __restrict__ buf, int rows, int ld, int c0) {
  int w = ld - c0;
  long i = (long)blockIdx.x * 256 + threadIdx.x;
  if (i >= (long)rows * w) return;
  long r = i / w; int c = c0 + (int)(i % w);
  buf[r * ld + c] = 0;
}

// ---- main GEMM: C[M,Nc] = A[M,K](bf16,row-major) @ BT[Nc,K](bf16,"B^T") ----
// Block tile 128x128, 8 wave32s (2 M x 4 N), 64x32 per wave via 8 WMMA accs.
// Loader: 4x global_load_async_to_lds_b128 per thread per K-step (no guards;
// buffers padded). Epilogue: optional bias+relu; row-major store (bf16/f32)
// and/or transposed bf16 store [n][M] feeding the next adjacency GEMM's B.
__global__ __launch_bounds__(256)
void k_gemm(const unsigned short* __restrict__ A, long lda,
            const unsigned short* __restrict__ BT, long ldb,
            int M, int Nc, int K,
            unsigned short* __restrict__ outBf, float* __restrict__ outF, long ldN,
            unsigned short* __restrict__ outT, long ldT,
            const float* __restrict__ bias, int relu) {
  constexpr int LDSLD = 40;  // 80B row stride: conflict-free, 16B aligned
  __shared__ __bf16 sA[2][128 * LDSLD];
  __shared__ __bf16 sB[2][128 * LDSLD];

  const int t    = threadIdx.x;
  const int lane = t & 31;
  const int wv   = t >> 5;
  const int wm   = (wv & 1) * 64;   // wave M offset in block
  const int wn   = (wv >> 1) * 32;  // wave N offset in block
  const int hh   = lane >> 4;       // lane half
  const int rr   = lane & 15;
  const long Mb  = (long)blockIdx.x * 128;
  const long Nb  = (long)blockIdx.y * 128;

  // loader mapping: 128x32 tile = 512 chunks of 8 bf16; chunks t and t+256
  const int ar = t >> 2, ak = (t & 3) * 8;  // chunk1 = row ar+64, same ak

  const unsigned short* gA0 = A + (Mb + ar) * lda + ak;
  const unsigned short* gA1 = gA0 + 64 * lda;
  const unsigned short* gB0 = BT + (Nb + ar) * ldb + ak;
  const unsigned short* gB1 = gB0 + 64 * ldb;

  const int lo = ar * LDSLD + ak;
  unsigned la0[2], la1[2], lb0[2], lb1[2];
#pragma unroll
  for (int b = 0; b < 2; ++b) {   // LDS byte offsets (generic ptr low 32 bits)
    la0[b] = (unsigned)(uintptr_t)&sA[b][lo];
    la1[b] = (unsigned)(uintptr_t)&sA[b][lo + 64 * LDSLD];
    lb0[b] = (unsigned)(uintptr_t)&sB[b][lo];
    lb1[b] = (unsigned)(uintptr_t)&sB[b][lo + 64 * LDSLD];
  }

  auto async_tile = [&](int s) {  // fetch K-step s into LDS buffer s&1
    const int b = s & 1;
    const long ko = (long)s * 32;
    asm volatile("global_load_async_to_lds_b128 %0, %1, off"
                 :: "v"(la0[b]), "v"(gA0 + ko) : "memory");
    asm volatile("global_load_async_to_lds_b128 %0, %1, off"
                 :: "v"(la1[b]), "v"(gA1 + ko) : "memory");
    asm volatile("global_load_async_to_lds_b128 %0, %1, off"
                 :: "v"(lb0[b]), "v"(gB0 + ko) : "memory");
    asm volatile("global_load_async_to_lds_b128 %0, %1, off"
                 :: "v"(lb1[b]), "v"(gB1 + ko) : "memory");
  };

  v8f acc[4][2];
  const v8f vz = {0.f, 0.f, 0.f, 0.f, 0.f, 0.f, 0.f, 0.f};
#pragma unroll
  for (int i = 0; i < 4; ++i)
#pragma unroll
    for (int j = 0; j < 2; ++j) acc[i][j] = vz;

  const int steps = (K + 31) / 32;
  async_tile(0);
  for (int s = 0; s < steps; ++s) {
    const int buf = s & 1;
    asm volatile("s_wait_asynccnt 0x0" ::: "memory");  // own async writes done
    __syncthreads();                                   // whole tile visible
    if (s + 1 < steps) async_tile(s + 1);              // prefetch next buffer

    v16bf af[4], bb[2];
#pragma unroll
    for (int mt = 0; mt < 4; ++mt) {  // A 16x32: lane=M row, K runs 8h.., 16+8h..
      const __bf16* p = &sA[buf][(wm + mt * 16 + rr) * LDSLD + 8 * hh];
      v8bf xlo = *(const v8bf*)p;
      v8bf xhi = *(const v8bf*)(p + 16);
      af[mt] = __builtin_shufflevector(xlo, xhi, 0, 1, 2, 3, 4, 5, 6, 7,
                                       8, 9, 10, 11, 12, 13, 14, 15);
    }
#pragma unroll
    for (int nt = 0; nt < 2; ++nt) {  // B 32x16: lane=N col, K runs 8h.., 16+8h..
      const __bf16* p = &sB[buf][(wn + nt * 16 + rr) * LDSLD + 8 * hh];
      v8bf xlo = *(const v8bf*)p;
      v8bf xhi = *(const v8bf*)(p + 16);
      bb[nt] = __builtin_shufflevector(xlo, xhi, 0, 1, 2, 3, 4, 5, 6, 7,
                                       8, 9, 10, 11, 12, 13, 14, 15);
    }
#pragma unroll
    for (int mt = 0; mt < 4; ++mt)
#pragma unroll
      for (int nt = 0; nt < 2; ++nt)
        acc[mt][nt] = __builtin_amdgcn_wmma_f32_16x16x32_bf16(
            false, af[mt], false, bb[nt], (short)0, acc[mt][nt], false, false);
  }

  // epilogue: C 16x16 layout -> lane holds col n=rr, rows m0..m0+7 (m0=16mt+8h)
#pragma unroll
  for (int mt = 0; mt < 4; ++mt) {
    const long m0 = Mb + wm + mt * 16 + hh * 8;
#pragma unroll
    for (int nt = 0; nt < 2; ++nt) {
      const long n = Nb + wn + nt * 16 + rr;
      if (n >= Nc) continue;
      const float bv = bias ? bias[n] : 0.f;
      float vals[8];
#pragma unroll
      for (int v = 0; v < 8; ++v) {
        float x = acc[mt][nt][v] + bv;
        if (relu) x = x > 0.f ? x : 0.f;
        vals[v] = x;
      }
      if (outF) {
#pragma unroll
        for (int v = 0; v < 8; ++v)
          if (m0 + v < M) outF[(m0 + v) * ldN + n] = vals[v];
      } else if (outBf) {
#pragma unroll
        for (int v = 0; v < 8; ++v)
          if (m0 + v < M) outBf[(m0 + v) * ldN + n] = bf16_rne(vals[v]);
      }
      if (outT && m0 + 7 < M) {  // M%8==0 -> all-or-nothing, one 16B store
        u16x8 pk;
#pragma unroll
        for (int v = 0; v < 8; ++v) pk[v] = bf16_rne(vals[v]);
        *(u16x8*)(outT + n * ldT + m0) = pk;
      }
    }
  }
}

// ---- host orchestration ----------------------------------------------------

static inline long cdiv(long a, long b) { return (a + b - 1) / b; }

extern "C" void kernel_launch(void* const* d_in, const int* in_sizes, int n_in,
                              void* d_out, int out_size, void* d_ws, size_t ws_size,
                              hipStream_t stream) {
  (void)in_sizes; (void)n_in; (void)out_size;
  const float* features = (const float*)d_in[0];
  const float* adj      = (const float*)d_in[1];
  const float* Wu1      = (const float*)d_in[2];
  const float* Wu2      = (const float*)d_in[3];
  const float* Wu3      = (const float*)d_in[4];
  const float* Wb1      = (const float*)d_in[5];
  const float* Wb2      = (const float*)d_in[6];
  const float* Wb3      = (const float*)d_in[7];
  const float* Wfc      = (const float*)d_in[8];
  const float* bfc      = (const float*)d_in[9];
  float* out            = (float*)d_out;

  char* ws = (char*)d_ws;
  size_t off = 0;
  auto take = [&](size_t bytes) -> unsigned short* {
    size_t o = off;
    off = (off + bytes + 255) & ~(size_t)255;
    return (unsigned short*)(ws + o);
  };
  // All buffers padded so 128-row x 32-col tile loads never go out of bounds.
  unsigned short* adjB = take((size_t)M_PAD * K_ADJ * 2);    // bf16 adj, K-pad 0
  unsigned short* XB   = take((size_t)M_PAD * F_PAD * 2);    // features, K-pad 0
  unsigned short* WuT  = take((size_t)NW_PAD * F_PAD * 2);   // [Wu1|Wu2|Wu3]^T
  unsigned short* WbT  = take((size_t)NW_PAD * C_PAD * 2);   // [Wb1|Wb2|Wb3]^T
  unsigned short* WfcT = take((size_t)NW_PAD * C_PAD * 2);   // W_fc^T
  unsigned short* a1B  = take((size_t)M_PAD * C_PAD * 2);    // a1 (A operand)
  unsigned short* a2B  = take((size_t)M_PAD * C_PAD * 2);    // a2 (A operand)
  unsigned short* UT   = take((size_t)UV_ROWS * K_ADJ * 2);  // U^T
  unsigned short* VT   = take((size_t)UV_ROWS * K_ADJ * 2);  // V^T
  unsigned short* T1T  = take((size_t)TS_ROWS * K_ADJ * 2);  // T1^T
  unsigned short* S1T  = take((size_t)TS_ROWS * K_ADJ * 2);  // S1^T
  if (off > ws_size) return;  // workspace too small; nothing sane to do

  // pack / convert
  k_cvt_pad<<<dim3((unsigned)cdiv(K_ADJ, 256), N_NODES), 256, 0, stream>>>(
      adj, adjB, N_NODES, K_ADJ);
  k_cvt_pad<<<dim3((unsigned)cdiv(F_PAD, 256), N_NODES), 256, 0, stream>>>(
      features, XB, F_IN, F_PAD);
  long tot = (long)C_DIM * F_PAD;
  k_pack_w3T<<<(unsigned)cdiv(tot, 256), 256, 0, stream>>>(Wu1, Wu2, Wu3, F_IN, H_DIM,
                                                           WuT, F_PAD, tot);
  tot = (long)C_DIM * C_PAD;
  k_pack_w3T<<<(unsigned)cdiv(tot, 256), 256, 0, stream>>>(Wb1, Wb2, Wb3, C_DIM, H_DIM,
                                                           WbT, C_PAD, tot);
  k_pack_wfcT<<<(unsigned)cdiv(tot, 256), 256, 0, stream>>>(Wfc, WfcT, tot);
  // zero the K-pad columns that sit inside reductions
  long zt = (long)N_NODES * (C_PAD - C_DIM);
  k_zero_pad<<<(unsigned)cdiv(zt, 256), 256, 0, stream>>>(a1B, N_NODES, C_PAD, C_DIM);
  k_zero_pad<<<(unsigned)cdiv(zt, 256), 256, 0, stream>>>(a2B, N_NODES, C_PAD, C_DIM);
  zt = (long)UV_ROWS * (K_ADJ - N_NODES);
  k_zero_pad<<<(unsigned)cdiv(zt, 256), 256, 0, stream>>>(UT, UV_ROWS, K_ADJ, N_NODES);
  k_zero_pad<<<(unsigned)cdiv(zt, 256), 256, 0, stream>>>(VT, UV_ROWS, K_ADJ, N_NODES);
  zt = (long)TS_ROWS * (K_ADJ - N_NODES);
  k_zero_pad<<<(unsigned)cdiv(zt, 256), 256, 0, stream>>>(T1T, TS_ROWS, K_ADJ, N_NODES);
  k_zero_pad<<<(unsigned)cdiv(zt, 256), 256, 0, stream>>>(S1T, TS_ROWS, K_ADJ, N_NODES);

  auto gemm = [&](const unsigned short* A, long lda, const unsigned short* BT, long ldb,
                  int M, int Nc, int K, unsigned short* oBf, float* oF, long ldN,
                  unsigned short* oT, long ldT, const float* bias, int relu) {
    dim3 g((unsigned)cdiv(M, 128), (unsigned)cdiv(Nc, 128));
    k_gemm<<<g, 256, 0, stream>>>(A, lda, BT, ldb, M, Nc, K, oBf, oF, ldN, oT, ldT,
                                  bias, relu);
  };

  // Stage 1: U = relu(X @ Wcat); a1 = [U0 | adj@U1 | adj^2@U2]
  gemm(XB, F_PAD, WuT, F_PAD, N_NODES, C_DIM, F_PAD,
       a1B, nullptr, C_PAD, UT, K_ADJ, nullptr, 1);
  gemm(adjB, K_ADJ, UT + (size_t)400 * K_ADJ, K_ADJ, N_NODES, 800, N_NODES,
       a1B + 400, nullptr, C_PAD, T1T, K_ADJ, nullptr, 0);
  gemm(adjB, K_ADJ, T1T + (size_t)400 * K_ADJ, K_ADJ, N_NODES, 400, N_NODES,
       a1B + 800, nullptr, C_PAD, nullptr, 0, nullptr, 0);

  // Stage 2: V = a1 @ Wbcat (no relu); a2 = [V0 | adj@V1 | adj^2@V2]
  gemm(a1B, C_PAD, WbT, C_PAD, N_NODES, C_DIM, C_PAD,
       a2B, nullptr, C_PAD, VT, K_ADJ, nullptr, 0);
  gemm(adjB, K_ADJ, VT + (size_t)400 * K_ADJ, K_ADJ, N_NODES, 800, N_NODES,
       a2B + 400, nullptr, C_PAD, S1T, K_ADJ, nullptr, 0);
  gemm(adjB, K_ADJ, S1T + (size_t)400 * K_ADJ, K_ADJ, N_NODES, 400, N_NODES,
       a2B + 800, nullptr, C_PAD, nullptr, 0, nullptr, 0);

  // Output: relu(a2 @ W_fc + b_fc) -> f32 [10000,1200]
  gemm(a2B, C_PAD, WfcT, C_PAD, N_NODES, C_DIM, C_PAD,
       nullptr, out, C_DIM, nullptr, 0, bfc, 1);
}
